// AttentionPooling_19146964206179
// MI455X (gfx1250) — matrix operand order repaired and checked by reading, not compiled, not run on hardware
//
#include <hip/hip_runtime.h>

// ---------------------------------------------------------------------------
// Attention pooling, fused flash-style single pass over x (1 GiB read once).
//   x:[32,8192,1024]f32  query:[16,1024]f32  proj_w:[1024,1024]f32  proj_b:[1024]f32
//   out:[32,1024]f32
// ---------------------------------------------------------------------------

#define BATCH  32
#define SEQ    8192
#define CH     1024
#define HEADS  16
#define NSPLIT 8
#define CHUNK  (SEQ / NSPLIT)   // 1024 rows per workgroup
#define TT     32               // seq rows per tile iteration
#define NW     8                // waves per 256-thread block (wave32)

typedef __attribute__((ext_vector_type(16))) __bf16 v16bf;
typedef __attribute__((ext_vector_type(2)))  __bf16 v2bf;
typedef __attribute__((ext_vector_type(8)))  float  v8f;

union ABf { v16bf v; unsigned int u[8]; };

// Packed fp32 -> 2 x bf16 (round-to-nearest-even), one VALU op when the
// packed convert builtin is available, else scalar v_cvt_bf16_f32 pair.
__device__ __forceinline__ unsigned int pkcvt(float a, float b) {
#if __has_builtin(__builtin_amdgcn_cvt_pk_bf16_f32)
  union { v2bf v; unsigned int u; } t;
  t.v = __builtin_amdgcn_cvt_pk_bf16_f32(a, b);
  return t.u;
#else
  union { __bf16 h[2]; unsigned int u; } t;
  t.h[0] = (__bf16)a; t.h[1] = (__bf16)b;
  return t.u;
#endif
}

// K-offset held by VGPR j of a 16x32 bf16 A-fragment (contiguous, matches B):
//   lanes 0-15: j0..7 -> K 0,2,...,14   (pairs K,K+1)
//   lanes 16-31: +16
__device__ __forceinline__ int ka(int j, int half) {
  return (j * 2) + 16 * half;
}

// ---------------------------------------------------------------------------
// Kernel 1: fused scores + online softmax + weighted pooling, split over seq.
// Writes un-normalized per-split partials: acc[b,s,h,c], m[b,s,h], l[b,s,h].
// ---------------------------------------------------------------------------
__global__ __launch_bounds__(256, 1)
void k_flash(const float* __restrict__ x, const float* __restrict__ q,
             float* __restrict__ part_acc, float* __restrict__ part_m,
             float* __restrict__ part_l) {
  const int b = blockIdx.x, split = blockIdx.y;
  const int tid  = threadIdx.x;
  const int wv   = tid >> 5, lane = tid & 31;
  const int half = lane >> 4, l16 = lane & 15;
  const int cw   = wv * 128;                 // this wave's channel slice

  __shared__ unsigned short x_tc[TT * CH];       // x tile, [t][c] bf16 (64KB)
  __shared__ unsigned short x_ct[CH * TT];       // x tile, [c][t] bf16 (64KB)
  __shared__ float s_part[NW][TT][HEADS];        // per-wave partial scores (16KB)
  __shared__ float s_full[TT][HEADS];            // reduced scores (2KB)
  __shared__ unsigned short p_tile[HEADS * TT];  // exp(s-m) bf16, [h][t] (1KB)
  __shared__ float corr_lds[HEADS];

  // Persistent B fragments of (q^T * scale): 32c x 16h per K-step of 32.
  // B layout: lane column n = l16 (head), lanes<16 hold K 0..15, lanes>=16 K 16..31,
  // VGPR j holds the pair (K=koff+2j, +1), contiguous in c of q[h][:].
  ABf qf[4];
  const float scale = 0.03125f;              // 1024^-0.5
#pragma unroll
  for (int kk = 0; kk < 4; ++kk)
#pragma unroll
    for (int j = 0; j < 8; ++j) {
      int k = cw + kk * 32 + half * 16 + 2 * j;
      qf[kk].u[j] = pkcvt(q[l16 * CH + k] * scale, q[l16 * CH + k + 1] * scale);
    }

  // Persistent accumulators: 8 WMMA tiles (16h x 16c) covering 128 channels.
  v8f acc[8];
#pragma unroll
  for (int i = 0; i < 8; ++i)
#pragma unroll
    for (int r = 0; r < 8; ++r) acc[i][r] = 0.f;

  float m_h = -1e30f, l_h = 0.f;             // per-head stats (wave0; lane = head
                                             // in each half-wave, kept coherent)

  const int r_stage = tid >> 3;              // 32 rows, 8 threads per row
  const int c_stage = (tid & 7) * 128;

  for (int it = 0; it < CHUNK / TT; ++it) {
    const int t0 = split * CHUNK + it * TT;

    // ---- stage x tile to LDS in both layouts (bf16) ----
    const float* src = x + ((size_t)b * SEQ + t0 + r_stage) * CH + c_stage;
    if (it + 1 < CHUNK / TT) {               // prefetch next tile's rows
      const float* nsrc = src + (size_t)TT * CH;
      __builtin_prefetch(nsrc, 0, 3);
      __builtin_prefetch(nsrc + 32, 0, 3);
      __builtin_prefetch(nsrc + 64, 0, 3);
      __builtin_prefetch(nsrc + 96, 0, 3);
    }
#pragma unroll 4
    for (int i = 0; i < 32; ++i) {
      float4 v = ((const float4*)src)[i];
      int c = c_stage + i * 4;
      unsigned int u01 = pkcvt(v.x, v.y);
      unsigned int u23 = pkcvt(v.z, v.w);
      ((unsigned int*)x_tc)[(r_stage * CH + c) >> 1]       = u01;
      ((unsigned int*)x_tc)[((r_stage * CH + c) >> 1) + 1] = u23;
      x_ct[(c + 0) * TT + r_stage] = (unsigned short)(u01 & 0xFFFFu);
      x_ct[(c + 1) * TT + r_stage] = (unsigned short)(u01 >> 16);
      x_ct[(c + 2) * TT + r_stage] = (unsigned short)(u23 & 0xFFFFu);
      x_ct[(c + 3) * TT + r_stage] = (unsigned short)(u23 >> 16);
    }
    __syncthreads();

    // ---- scores: partial over this wave's 128 channels, 2 subtiles of 16 t ----
#pragma unroll
    for (int sub = 0; sub < 2; ++sub) {
      v8f sc;
#pragma unroll
      for (int r = 0; r < 8; ++r) sc[r] = 0.f;
      const int row = sub * 16 + l16;        // A row = t within subtile
#pragma unroll
      for (int kk = 0; kk < 4; ++kk) {
        ABf af;
#pragma unroll
        for (int j = 0; j < 8; ++j) {
          int c = cw + kk * 32 + ka(j, half);
          af.u[j] = ((const unsigned int*)x_tc)[(row * CH + c) >> 1];
        }
        sc = __builtin_amdgcn_wmma_f32_16x16x32_bf16(false, af.v, false, qf[kk].v,
                                                     (short)0, sc, false, false);
      }
      // C layout: VGPR r -> row m = r + 8*half, col n = l16
#pragma unroll
      for (int r = 0; r < 8; ++r)
        s_part[wv][sub * 16 + r + 8 * half][l16] = sc[r];
    }
    __syncthreads();

    // ---- reduce channel-partial scores across the 8 waves ----
    for (int e = tid; e < TT * HEADS; e += 256) {
      int t = e >> 4, h = e & 15;
      float s = 0.f;
#pragma unroll
      for (int w = 0; w < NW; ++w) s += s_part[w][t][h];
      s_full[t][h] = s;
    }
    __syncthreads();

    // ---- online softmax update (wave 0; each half-wave owns 16 of 32 rows,
    //      merged with cross-half shuffles so both halves track m/l) ----
    if (wv == 0) {
      const int h = l16;
      const int tb = half * 16;
      float tmax = -1e30f;
#pragma unroll
      for (int t = 0; t < 16; ++t) tmax = fmaxf(tmax, s_full[tb + t][h]);
      tmax = fmaxf(tmax, __shfl_xor(tmax, 16));
      float mnew = fmaxf(m_h, tmax);
      float corr = __expf(m_h - mnew);
      float ps = 0.f;
#pragma unroll
      for (int t2 = 0; t2 < 8; ++t2) {
        float p0 = __expf(s_full[tb + 2 * t2][h] - mnew);
        float p1 = __expf(s_full[tb + 2 * t2 + 1][h] - mnew);
        ps += p0 + p1;
        ((unsigned int*)p_tile)[(h * TT + tb) / 2 + t2] = pkcvt(p0, p1);
      }
      ps += __shfl_xor(ps, 16);
      l_h = l_h * corr + ps;
      m_h = mnew;
      if (half == 0) corr_lds[h] = corr;
    }
    __syncthreads();

    // ---- acc = acc*corr + P(16h x 32t) x X(32t x 16c), per 16-channel chunk ----
    {
      float c8[8];
#pragma unroll
      for (int r = 0; r < 8; ++r) c8[r] = corr_lds[r + 8 * half];
      ABf pf;                                // A fragment of P, row = head
#pragma unroll
      for (int j = 0; j < 8; ++j)
        pf.u[j] = ((const unsigned int*)p_tile)[(l16 * TT + ka(j, half)) >> 1];
#pragma unroll
      for (int i = 0; i < 8; ++i) {
        ABf bf;                              // B fragment of X, col = channel
        const int c0 = cw + i * 16 + l16;
#pragma unroll
        for (int j = 0; j < 8; ++j) {
          int t = half * 16 + 2 * j;
          bf.u[j] = ((const unsigned int*)x_ct)[(c0 * TT + t) >> 1];
        }
        v8f a = acc[i];
#pragma unroll
        for (int r = 0; r < 8; ++r) a[r] *= c8[r];
        acc[i] = __builtin_amdgcn_wmma_f32_16x16x32_bf16(false, pf.v, false, bf.v,
                                                         (short)0, a, false, false);
      }
    }
    __syncthreads();  // protect LDS tiles before next stage
  }

  // ---- epilogue: write per-split partials ----
  const size_t base = (size_t)(b * NSPLIT + split) * HEADS;
#pragma unroll
  for (int i = 0; i < 8; ++i)
#pragma unroll
    for (int r = 0; r < 8; ++r)
      part_acc[(base + r + 8 * half) * CH + cw + i * 16 + l16] = acc[i][r];
  if (wv == 0 && lane < 16) {
    part_m[base + lane] = m_h;
    part_l[base + lane] = l_h;
  }
}

// ---------------------------------------------------------------------------
// Kernel 2: merge seq-splits (softmax rescale), normalize, head-mean.
// ---------------------------------------------------------------------------
__global__ __launch_bounds__(256, 1)
void k_combine(const float* __restrict__ part_acc, const float* __restrict__ part_m,
               const float* __restrict__ part_l, float* __restrict__ meanpool) {
  const int b = blockIdx.x, tid = threadIdx.x;
  __shared__ float Mh[HEADS], Lh[HEADS], Wsh[NSPLIT][HEADS];
  if (tid < HEADS) {
    float M = -1e30f;
    for (int s = 0; s < NSPLIT; ++s)
      M = fmaxf(M, part_m[(b * NSPLIT + s) * HEADS + tid]);
    float L = 0.f;
    for (int s = 0; s < NSPLIT; ++s)
      L += part_l[(b * NSPLIT + s) * HEADS + tid] *
           __expf(part_m[(b * NSPLIT + s) * HEADS + tid] - M);
    Mh[tid] = M; Lh[tid] = L;
  }
  __syncthreads();
  if (tid < NSPLIT * HEADS) {
    int s = tid >> 4, h = tid & 15;
    Wsh[s][h] = __expf(part_m[(b * NSPLIT + s) * HEADS + h] - Mh[h]) /
                ((float)HEADS * Lh[h]);
  }
  __syncthreads();
  for (int c = tid; c < CH; c += 256) {
    float sum = 0.f;
    for (int s = 0; s < NSPLIT; ++s)
#pragma unroll
      for (int h = 0; h < HEADS; ++h)
        sum += part_acc[((size_t)(b * NSPLIT + s) * HEADS + h) * CH + c] * Wsh[s][h];
    meanpool[b * CH + c] = sum;
  }
}

// ---------------------------------------------------------------------------
// Kernel 3: out[32,1024] = meanpool[32,1024] @ proj_w^T + proj_b  (WMMA GEMM)
// grid: 64 column tiles; waves split (2 row tiles) x (4-way K), LDS reduce.
// ---------------------------------------------------------------------------
__global__ __launch_bounds__(256, 1)
void k_proj(const float* __restrict__ mp, const float* __restrict__ pw,
            const float* __restrict__ pb, float* __restrict__ out) {
  const int tid = threadIdx.x, wv = tid >> 5, lane = tid & 31;
  const int half = lane >> 4, l16 = lane & 15;
  const int jbase = blockIdx.x * 16;
  const int mtile = wv & 1;
  const int kbase = (wv >> 1) * 256;
  __shared__ float part[NW][16][16];

  v8f acc;
#pragma unroll
  for (int r = 0; r < 8; ++r) acc[r] = 0.f;
  const int brow = mtile * 16 + l16;     // batch row (A row)
  const int jrow = jbase + l16;          // output column = proj_w row (B col)
#pragma unroll
  for (int kk = 0; kk < 8; ++kk) {
    ABf af, bf;
#pragma unroll
    for (int j = 0; j < 8; ++j) {
      int k = kbase + kk * 32 + ka(j, half);
      af.u[j] = pkcvt(mp[brow * CH + k], mp[brow * CH + k + 1]);
      int k2 = kbase + kk * 32 + half * 16 + 2 * j;
      bf.u[j] = pkcvt(pw[jrow * CH + k2], pw[jrow * CH + k2 + 1]);
    }
    acc = __builtin_amdgcn_wmma_f32_16x16x32_bf16(false, af.v, false, bf.v,
                                                  (short)0, acc, false, false);
  }
#pragma unroll
  for (int r = 0; r < 8; ++r) part[wv][r + 8 * half][l16] = acc[r];
  __syncthreads();
  for (int e = tid; e < 2 * 16 * 16; e += 256) {
    int mt = e >> 8, m = (e >> 4) & 15, n = e & 15;
    float s = 0.f;
#pragma unroll
    for (int w = 0; w < 4; ++w) s += part[2 * w + mt][m][n];
    out[(mt * 16 + m) * CH + jbase + n] = s + pb[jbase + n];
  }
}

// ---------------------------------------------------------------------------
extern "C" void kernel_launch(void* const* d_in, const int* in_sizes, int n_in,
                              void* d_out, int out_size, void* d_ws, size_t ws_size,
                              hipStream_t stream) {
  (void)in_sizes; (void)n_in; (void)out_size; (void)ws_size;
  const float* x  = (const float*)d_in[0];
  const float* q  = (const float*)d_in[1];
  const float* pw = (const float*)d_in[2];
  const float* pb = (const float*)d_in[3];
  float* out = (float*)d_out;

  // workspace layout (floats): 16 MiB partial acc + stats + meanpool (~16.2 MiB)
  float* ws       = (float*)d_ws;
  float* part_acc = ws;                                              // 32*8*16*1024
  float* part_m   = ws + (size_t)BATCH * NSPLIT * HEADS * CH;        // 4096
  float* part_l   = part_m + BATCH * NSPLIT * HEADS;                 // 4096
  float* meanpool = part_l + BATCH * NSPLIT * HEADS;                 // 32*1024

  k_flash  <<<dim3(BATCH, NSPLIT), 256, 0, stream>>>(x, q, part_acc, part_m, part_l);
  k_combine<<<BATCH,               256, 0, stream>>>(part_acc, part_m, part_l, meanpool);
  k_proj   <<<CH / 16,             256, 0, stream>>>(meanpool, pw, pb, out);
}
